// HIHI_90357521973479
// MI455X (gfx1250) — matrix operand order repaired and checked by reading, not compiled, not run on hardware
//
#include <hip/hip_runtime.h>

// Problem constants (from reference: B=16, C=512, H=W=28, K=2048)
#define Bn 16
#define Cc 512
#define Pp 784            // H*W
#define Nn 12544          // B*H*W
#define Kk 2048
#define NCe 6422528       // Nn*Cc (= one x output, elements)

#define MROWS 32          // rows per block (2 WMMA M-tiles)
#define LDSS  520         // padded LDS row stride in bf16 elems (512 + 8 -> bank-conflict-free)

typedef __attribute__((ext_vector_type(16))) __bf16 v16bf;
typedef __attribute__((ext_vector_type(8)))  __bf16 v8bf;
typedef __attribute__((ext_vector_type(8)))  float  v8f;
typedef __attribute__((ext_vector_type(4)))  unsigned int u32x4;
typedef __attribute__((ext_vector_type(8)))  int i32x8;
typedef __attribute__((ext_vector_type(4)))  int i32x4;

union Frag { v16bf v; v8bf h[2]; };

#if __has_builtin(__builtin_amdgcn_tensor_load_to_lds) && \
    __has_builtin(__builtin_amdgcn_s_wait_tensorcnt)
#define HAVE_TDM 1
#else
#define HAVE_TDM 0
#endif

// ---------------------------------------------------------------------------
// (B,C,H,W) -> (N,C) fp32, plus bf16 hi/lo split planes for compensated WMMA.
// ---------------------------------------------------------------------------
__global__ void k_transpose(const float* __restrict__ head, float* __restrict__ F,
                            __bf16* __restrict__ Fhi, __bf16* __restrict__ Flo) {
  int e = blockIdx.x * 256 + threadIdx.x;   // e < Nn*Cc
  int n = e >> 9;                           // /Cc
  int c = e & 511;
  int b = n / Pp, p = n % Pp;
  float x = head[(size_t)b * (Cc * Pp) + (size_t)c * Pp + p];
  F[e] = x;
  __bf16 h = (__bf16)x;
  Fhi[e] = h;
  Flo[e] = (__bf16)(x - (float)h);
}

// ---------------------------------------------------------------------------
// Per codebook row: squared norm + bf16 hi/lo planes. 1 block / row.
// ---------------------------------------------------------------------------
__global__ void k_prep_cb(const float* __restrict__ cb, __bf16* __restrict__ hi,
                          __bf16* __restrict__ lo, float* __restrict__ norm) {
  int k = blockIdx.x;
  const float* r = cb + (size_t)k * Cc;
  float acc = 0.f;
  for (int c = threadIdx.x; c < Cc; c += 256) {
    float x = r[c];
    acc += x * x;
    __bf16 h = (__bf16)x;
    hi[(size_t)k * Cc + c] = h;
    lo[(size_t)k * Cc + c] = (__bf16)(x - (float)h);
  }
  for (int o = 16; o; o >>= 1) acc += __shfl_down(acc, o, 32);
  __shared__ float red[8];
  int w = threadIdx.x >> 5, ln = threadIdx.x & 31;
  if (ln == 0) red[w] = acc;
  __syncthreads();
  if (threadIdx.x == 0) {
    float t = 0.f;
    for (int i = 0; i < 8; ++i) t += red[i];
    norm[k] = t;
  }
}

// ---------------------------------------------------------------------------
// Fused GEMM (bf16x3 compensated WMMA) + row argmin.
// Block = 32 rows (2 M-tiles) staged in LDS via TDM; 8 waves stride over the
// 128 code-tiles. argmin key: ||code||^2 - 2*(f . code).
// ---------------------------------------------------------------------------
__global__ __launch_bounds__(256) void k_gemm_argmin(
    const __bf16* __restrict__ Fhi, const __bf16* __restrict__ Flo,
    const __bf16* __restrict__ Chi, const __bf16* __restrict__ Clo,
    const float* __restrict__ cbnorm, int* __restrict__ idxout) {
  const int row0 = blockIdx.x * MROWS;
  const int wave = threadIdx.x >> 5;
  const int lane = threadIdx.x & 31;
  const int l15  = lane & 15;
  const int sel  = lane >> 4;

  __shared__ __bf16 sAhi[MROWS * LDSS];   // 33,280 B
  __shared__ __bf16 sAlo[MROWS * LDSS];   // 33,280 B
  __shared__ float s_sc[8][MROWS];
  __shared__ int   s_ix[8][MROWS];

#if HAVE_TDM
  if (threadIdx.x < 32) {
    // 2D TDM tile: 32 rows x 512 bf16, row stride 512 elems, +16B LDS pad/row.
    i32x8 g1;
    g1[0] = (1 << 16)            // data_size = 2B
          | (1 << 20)            // pad_enable
          | (7 << 22)            // pad_interval: every 256 DWORDs (= 1 row)
          | (3 << 25);           // pad_amount: 4 DWORDs (= 16 B)
    g1[1] = (int)(512u << 16);   // tensor_dim0 = 512 (elems)
    g1[2] = (int)((unsigned)MROWS << 16);  // tensor_dim1 = 32
    g1[3] = (int)(512u << 16);   // tile_dim0 = 512
    g1[4] = MROWS;               // tile_dim1 = 32
    g1[5] = 512;                 // tensor_dim0_stride = 512 (elems)
    g1[6] = 0;
    g1[7] = 0;
    i32x4 gz4 = {0, 0, 0, 0};                      // groups 2/3: NULL (2D tensor)
    i32x8 gz8 = {0, 0, 0, 0, 0, 0, 0, 0};          // 6-arg form: trailing group, zeroed

    unsigned long long gah =
        (unsigned long long)(uintptr_t)(Fhi + (size_t)row0 * Cc);
    unsigned long long gal =
        (unsigned long long)(uintptr_t)(Flo + (size_t)row0 * Cc);
    u32x4 g0h, g0l;
    g0h[0] = 1u;                                      // count=1 (valid D#)
    g0h[1] = (unsigned)(uintptr_t)sAhi;               // LDS byte address
    g0h[2] = (unsigned)(gah & 0xFFFFFFFFu);
    g0h[3] = (unsigned)((gah >> 32) & 0x1FFFFFFu) | (2u << 30);  // type=2
    g0l[0] = 1u;
    g0l[1] = (unsigned)(uintptr_t)sAlo;
    g0l[2] = (unsigned)(gal & 0xFFFFFFFFu);
    g0l[3] = (unsigned)((gal >> 32) & 0x1FFFFFFu) | (2u << 30);

    __builtin_amdgcn_tensor_load_to_lds(g0h, g1, gz4, gz4, gz8, 0);
    __builtin_amdgcn_tensor_load_to_lds(g0l, g1, gz4, gz4, gz8, 0);
    __builtin_amdgcn_s_wait_tensorcnt(0);
  }
#else
  // Cooperative fallback fill (padded layout), 16B chunks.
  for (int i = threadIdx.x; i < MROWS * 64; i += 256) {
    int r = i >> 6, c4 = i & 63;
    *((u32x4*)(sAhi + r * LDSS) + c4) =
        *((const u32x4*)(Fhi + (size_t)(row0 + r) * Cc) + c4);
    *((u32x4*)(sAlo + r * LDSS) + c4) =
        *((const u32x4*)(Flo + (size_t)(row0 + r) * Cc) + c4);
  }
#endif
  __syncthreads();

  // A fragment rows in LDS (tile0 = rows 0..15, tile1 = rows 16..31)
  const __bf16* a0h = sAhi + (size_t)l15 * LDSS;
  const __bf16* a0l = sAlo + (size_t)l15 * LDSS;
  const __bf16* a1h = sAhi + (size_t)(16 + l15) * LDSS;
  const __bf16* a1l = sAlo + (size_t)(16 + l15) * LDSS;

  float best0[8], best1[8];
  int   bidx0[8], bidx1[8];
#pragma unroll
  for (int j = 0; j < 8; ++j) {
    best0[j] = 3.4e38f; bidx0[j] = 0;
    best1[j] = 3.4e38f; bidx1[j] = 0;
  }

  for (int kt = wave; kt < Kk / 16; kt += 8) {
    const int code = kt * 16 + l15;   // B column n = lane (mod 16)
    const __bf16* bHiRow = Chi + (size_t)code * Cc + sel * 8;
    const __bf16* bLoRow = Clo + (size_t)code * Cc + sel * 8;

    v8f acc0 = {}, acc1 = {};
#pragma unroll 4
    for (int c0 = 0; c0 < Cc; c0 += 32) {
      Frag bh, bl, a0H, a0L, a1H, a1L;
      bh.h[0] = *(const v8bf*)(bHiRow + c0);
      bh.h[1] = *(const v8bf*)(bHiRow + c0 + 16);
      bl.h[0] = *(const v8bf*)(bLoRow + c0);
      bl.h[1] = *(const v8bf*)(bLoRow + c0 + 16);
      a0H.h[0] = *(const v8bf*)(a0h + c0 + sel * 8);
      a0H.h[1] = *(const v8bf*)(a0h + c0 + 16 + sel * 8);
      a0L.h[0] = *(const v8bf*)(a0l + c0 + sel * 8);
      a0L.h[1] = *(const v8bf*)(a0l + c0 + 16 + sel * 8);
      a1H.h[0] = *(const v8bf*)(a1h + c0 + sel * 8);
      a1H.h[1] = *(const v8bf*)(a1h + c0 + 16 + sel * 8);
      a1L.h[0] = *(const v8bf*)(a1l + c0 + sel * 8);
      a1L.h[1] = *(const v8bf*)(a1l + c0 + 16 + sel * 8);
      // compensated fp32-accuracy: hi*hi + hi*lo + lo*hi
      acc0 = __builtin_amdgcn_wmma_f32_16x16x32_bf16(false, a0H.v, false, bh.v,
                                                     (short)0, acc0, false, false);
      acc0 = __builtin_amdgcn_wmma_f32_16x16x32_bf16(false, a0H.v, false, bl.v,
                                                     (short)0, acc0, false, false);
      acc0 = __builtin_amdgcn_wmma_f32_16x16x32_bf16(false, a0L.v, false, bh.v,
                                                     (short)0, acc0, false, false);
      acc1 = __builtin_amdgcn_wmma_f32_16x16x32_bf16(false, a1H.v, false, bh.v,
                                                     (short)0, acc1, false, false);
      acc1 = __builtin_amdgcn_wmma_f32_16x16x32_bf16(false, a1H.v, false, bl.v,
                                                     (short)0, acc1, false, false);
      acc1 = __builtin_amdgcn_wmma_f32_16x16x32_bf16(false, a1L.v, false, bh.v,
                                                     (short)0, acc1, false, false);
    }
    const float cn = cbnorm[code];
#pragma unroll
    for (int j = 0; j < 8; ++j) {   // D elem j -> row (sel*8 + j), col l15
      float s0 = cn - 2.0f * acc0[j];
      if (s0 < best0[j] || (s0 == best0[j] && code < bidx0[j])) {
        best0[j] = s0; bidx0[j] = code;
      }
      float s1 = cn - 2.0f * acc1[j];
      if (s1 < best1[j] || (s1 == best1[j] && code < bidx1[j])) {
        best1[j] = s1; bidx1[j] = code;
      }
    }
  }

  // reduce over the 16 lanes of each half (columns of the tile)
#pragma unroll
  for (int off = 1; off < 16; off <<= 1) {
#pragma unroll
    for (int j = 0; j < 8; ++j) {
      float os = __shfl_xor(best0[j], off, 32);
      int   oi = __shfl_xor(bidx0[j], off, 32);
      if (os < best0[j] || (os == best0[j] && oi < bidx0[j])) {
        best0[j] = os; bidx0[j] = oi;
      }
      os = __shfl_xor(best1[j], off, 32);
      oi = __shfl_xor(bidx1[j], off, 32);
      if (os < best1[j] || (os == best1[j] && oi < bidx1[j])) {
        best1[j] = os; bidx1[j] = oi;
      }
    }
  }

  if (l15 == 0) {
#pragma unroll
    for (int j = 0; j < 8; ++j) {
      s_sc[wave][sel * 8 + j]      = best0[j];
      s_ix[wave][sel * 8 + j]      = bidx0[j];
      s_sc[wave][16 + sel * 8 + j] = best1[j];
      s_ix[wave][16 + sel * 8 + j] = bidx1[j];
    }
  }
  __syncthreads();
  if (threadIdx.x < MROWS) {        // reduce across the 8 waves
    float bs = 3.4e38f; int bi = 0x7fffffff;
    for (int w = 0; w < 8; ++w) {
      float s = s_sc[w][threadIdx.x]; int i2 = s_ix[w][threadIdx.x];
      if (s < bs || (s == bs && i2 < bi)) { bs = s; bi = i2; }
    }
    idxout[row0 + threadIdx.x] = bi;
  }
}

// ---------------------------------------------------------------------------
// Gather q = cb[idx], emit x_s and recon (back-transposed), residual in place
// (+ refreshed bf16 planes), block-reduced sum(r^2) for the loss.
// ---------------------------------------------------------------------------
__global__ void k_update(const int* __restrict__ idx, const float* __restrict__ cb,
                         float* __restrict__ F, __bf16* __restrict__ Fhi,
                         __bf16* __restrict__ Flo, float* __restrict__ xout,
                         float* __restrict__ recon, int first,
                         float* __restrict__ lossacc) {
  int e = blockIdx.x * 256 + threadIdx.x;   // e < Nn*Cc
  int n = e >> 9, c = e & 511;
  float q = cb[(size_t)idx[n] * Cc + c];
  float f = F[e];
  float r = f - q;
  int b = n / Pp, p = n % Pp;
  size_t oi = (size_t)b * (Cc * Pp) + (size_t)c * Pp + p;
  xout[oi] = q;
  if (first) recon[oi] = q; else recon[oi] += q;
  F[e] = r;
  __bf16 h = (__bf16)r;
  Fhi[e] = h;
  Flo[e] = (__bf16)(r - (float)h);

  float v = r * r;
  for (int o = 16; o; o >>= 1) v += __shfl_down(v, o, 32);
  __shared__ float red[8];
  int w = threadIdx.x >> 5, ln = threadIdx.x & 31;
  if (ln == 0) red[w] = v;
  __syncthreads();
  if (threadIdx.x == 0) {
    float t = 0.f;
    for (int i = 0; i < 8; ++i) t += red[i];
    atomicAdd(lossacc, t);
  }
}

__global__ void k_init_loss(float* loss) {
  if (threadIdx.x < 3) loss[threadIdx.x] = 0.f;
}

// loss = q_loss + e_loss = 2 * mean(r^2)
__global__ void k_final_loss(const float* __restrict__ loss, float* __restrict__ out) {
  if (threadIdx.x < 3) out[threadIdx.x] = 2.0f * loss[threadIdx.x] / (float)NCe;
}

// ---------------------------------------------------------------------------
extern "C" void kernel_launch(void* const* d_in, const int* in_sizes, int n_in,
                              void* d_out, int out_size, void* d_ws, size_t ws_size,
                              hipStream_t stream) {
  const float* head = (const float*)d_in[0];
  const float* cbs[3] = {(const float*)d_in[1], (const float*)d_in[2],
                         (const float*)d_in[3]};
  float* out = (float*)d_out;
  char*  ws  = (char*)d_ws;

  // workspace carve-up (~64 MB total)
  size_t off = 0;
  float*  F    = (float*)(ws + off);  off += (size_t)NCe * 4;
  __bf16* Fhi  = (__bf16*)(ws + off); off += (size_t)NCe * 2;
  __bf16* Flo  = (__bf16*)(ws + off); off += (size_t)NCe * 2;
  __bf16* Chi  = (__bf16*)(ws + off); off += (size_t)3 * Kk * Cc * 2;
  __bf16* Clo  = (__bf16*)(ws + off); off += (size_t)3 * Kk * Cc * 2;
  float*  cbn  = (float*)(ws + off);  off += (size_t)3 * Kk * 4;
  int*    idxb = (int*)(ws + off);    off += (size_t)Nn * 4;
  float*  lacc = (float*)(ws + off);

  // output layout: x1 | x2 | x3 | loss1 loss2 loss3 | recon
  float* xouts[3] = {out, out + (size_t)NCe, out + 2 * (size_t)NCe};
  float* lossout  = out + 3 * (size_t)NCe;
  float* recon    = out + 3 * (size_t)NCe + 3;

  k_init_loss<<<1, 32, 0, stream>>>(lacc);
  k_transpose<<<NCe / 256, 256, 0, stream>>>(head, F, Fhi, Flo);
  for (int s = 0; s < 3; ++s)
    k_prep_cb<<<Kk, 256, 0, stream>>>(cbs[s], Chi + (size_t)s * Kk * Cc,
                                      Clo + (size_t)s * Kk * Cc, cbn + s * Kk);
  for (int s = 0; s < 3; ++s) {
    k_gemm_argmin<<<Nn / MROWS, 256, 0, stream>>>(
        Fhi, Flo, Chi + (size_t)s * Kk * Cc, Clo + (size_t)s * Kk * Cc,
        cbn + s * Kk, idxb);
    k_update<<<NCe / 256, 256, 0, stream>>>(idxb, cbs[s], F, Fhi, Flo,
                                            xouts[s], recon, s == 0, lacc + s);
  }
  k_final_loss<<<1, 32, 0, stream>>>(lacc, lossout);
}
